// SparseChannelMix_24163486007380
// MI455X (gfx1250) — compile-verified, matrix operand order checked by reading, not run on hardware
//
#include <hip/hip_runtime.h>
#include <hip/hip_bf16.h>

// Problem constants (match reference)
#define D_MODEL 1024
#define D_FF    4096
#define NROWS   8192   // B*T = 4*2048
#define TOPK    614

typedef __attribute__((ext_vector_type(16))) __bf16 v16bf;
typedef __attribute__((ext_vector_type(8)))  __bf16 v8bf;
typedef __attribute__((ext_vector_type(8)))  float  v8f;

__device__ __forceinline__ __bf16 f32_to_bf16(float f) {
  unsigned u = __builtin_bit_cast(unsigned, f);
  unsigned r = u + 0x7FFFu + ((u >> 16) & 1u);   // round-to-nearest-even
  unsigned short h = (unsigned short)(r >> 16);
  return __builtin_bit_cast(__bf16, h);
}

// One 16-element bf16 WMMA operand fragment for this lane: two 16-byte chunks
// at p and p+16 elements (16-bit A/B operand K layout).
__device__ __forceinline__ v16bf load_frag(const __bf16* p) {
  v8bf lo = *(const v8bf*)(p);
  v8bf hi = *(const v8bf*)(p + 16);
  return __builtin_shufflevector(lo, hi,
      0, 1, 2, 3, 4, 5, 6, 7, 8, 9, 10, 11, 12, 13, 14, 15);
}

__device__ __forceinline__ v8f wmma_bf16(v16bf a, v16bf b, v8f c) {
  return __builtin_amdgcn_wmma_f32_16x16x32_bf16(false, a, false, b,
                                                 (short)0, c, false, false);
}

// ---------------------------------------------------------------------------
// Kernel 0: f32 -> bf16 conversion (bandwidth-trivial).
// ---------------------------------------------------------------------------
__global__ __launch_bounds__(256)
void cvt_f32_bf16(const float* __restrict__ in, __bf16* __restrict__ out, int n) {
  int i = blockIdx.x * 256 + threadIdx.x;
  if (i < n) out[i] = f32_to_bf16(in[i]);
}

// ---------------------------------------------------------------------------
// Kernel 1: fused gate/up GEMM. One wave owns a 32(token) x 32(feature) tile:
// 2 A frags x (2 W1 + 2 W2) frags -> 8 WMMA per K-step from 6 fragment loads.
//   act[t][f] = silu(x @ W1^T)   (f32, for exact top-k)
//   hb [t][f] = act * (x @ W2^T) (bf16, feeds final GEMM)
// ---------------------------------------------------------------------------
__global__ __launch_bounds__(256)
void gemm_gate_act(const __bf16* __restrict__ xb,
                   const __bf16* __restrict__ w1b,
                   const __bf16* __restrict__ w2b,
                   float*  __restrict__ act,
                   __bf16* __restrict__ hb) {
  const int lane = threadIdx.x & 31;
  const int wave = threadIdx.x >> 5;
  const int m0 = blockIdx.x * 32;                 // token tile (32 rows)
  const int f0 = (blockIdx.y * 8 + wave) * 32;    // feature tile (32 cols)
  const int r  = lane & 15;
  const int kh = lane >> 4;

  const __bf16* a0  = xb  + (size_t)(m0 + r) * D_MODEL + kh * 8;
  const __bf16* a1  = a0  + (size_t)16 * D_MODEL;
  const __bf16* p1a = w1b + (size_t)(f0 + r) * D_MODEL + kh * 8;
  const __bf16* p1b = p1a + (size_t)16 * D_MODEL;
  const __bf16* p2a = w2b + (size_t)(f0 + r) * D_MODEL + kh * 8;
  const __bf16* p2b = p2a + (size_t)16 * D_MODEL;

  v8f c1[2][2] = {};   // x @ W1^T (pre-activation)
  v8f c2[2][2] = {};   // x @ W2^T (gate)
  for (int kk = 0; kk < D_MODEL; kk += 32) {
    __builtin_prefetch(p1a + kk + 128, 0, 1);
    __builtin_prefetch(p2a + kk + 128, 0, 1);
    v16bf A0  = load_frag(a0  + kk);
    v16bf A1  = load_frag(a1  + kk);
    v16bf B1a = load_frag(p1a + kk);
    v16bf B1b = load_frag(p1b + kk);
    v16bf B2a = load_frag(p2a + kk);
    v16bf B2b = load_frag(p2b + kk);
    c1[0][0] = wmma_bf16(A0, B1a, c1[0][0]);
    c1[0][1] = wmma_bf16(A0, B1b, c1[0][1]);
    c1[1][0] = wmma_bf16(A1, B1a, c1[1][0]);
    c1[1][1] = wmma_bf16(A1, B1b, c1[1][1]);
    c2[0][0] = wmma_bf16(A0, B2a, c2[0][0]);
    c2[0][1] = wmma_bf16(A0, B2b, c2[0][1]);
    c2[1][0] = wmma_bf16(A1, B2a, c2[1][0]);
    c2[1][1] = wmma_bf16(A1, B2b, c2[1][1]);
  }

  // C/D layout: col n = lane&15, row m = i + 8*(lane>>4) for VGPR i.
  #pragma unroll
  for (int mi = 0; mi < 2; ++mi) {
    #pragma unroll
    for (int fi = 0; fi < 2; ++fi) {
      #pragma unroll
      for (int i = 0; i < 8; ++i) {
        int m = m0 + mi * 16 + i + 8 * kh;
        int f = f0 + fi * 16 + r;
        float pre = c1[mi][fi][i];
        float s   = pre / (1.0f + __expf(-pre));   // SiLU
        size_t off = (size_t)m * D_FF + (size_t)f;
        act[off] = s;
        hb[off]  = f32_to_bf16(s * c2[mi][fi][i]);
      }
    }
  }
}

// ---------------------------------------------------------------------------
// Kernel 2: exact per-row top-k threshold via 4-pass MSB-first radix select.
// The 16 KB act row is staged memory->LDS with GLOBAL_LOAD_ASYNC_TO_LDS_B128
// (ASYNCcnt path), then selected in LDS. Dynamic LDS so per-lane LDS byte
// addresses are plain integers from base 0.
//   smem layout: [0,16384)  keys (4096 u32)
//                [16384, +1024) hist (256 int)
//                [17408, +8)    s_digit, s_k
// ---------------------------------------------------------------------------
#define TOPK_SMEM_BYTES (D_FF * 4 + 256 * 4 + 16)

__global__ __launch_bounds__(256)
void topk_mask(const float* __restrict__ act, __bf16* __restrict__ hb) {
  extern __shared__ unsigned char smem_raw[];
  unsigned* keys = (unsigned*)smem_raw;                    // LDS byte off 0
  int* hist      = (int*)(smem_raw + D_FF * 4);
  int* s_digit   = (int*)(smem_raw + D_FF * 4 + 1024);
  int* s_k       = s_digit + 1;

  const int row = blockIdx.x;
  const int tid = threadIdx.x;
  const float* arow = act + (size_t)row * D_FF;

  // Async-stage the row into LDS: 256 lanes x 16 B x 4 iters = 16 KB.
  #pragma unroll
  for (int it = 0; it < 4; ++it) {
    unsigned ldsa = (unsigned)(tid * 16 + it * 4096);
    const float* g = arow + it * 1024 + tid * 4;
    asm volatile("global_load_async_to_lds_b128 %0, %1, off"
                 :: "v"(ldsa), "v"(g) : "memory");
  }
  asm volatile("s_wait_asynccnt 0x0" ::: "memory");
  __syncthreads();

  // In-place monotone map: descending float == descending unsigned.
  for (int i = tid; i < D_FF; i += 256) {
    unsigned u = keys[i];
    keys[i] = (u & 0x80000000u) ? ~u : (u | 0x80000000u);
  }
  if (tid == 0) s_k[0] = TOPK;
  __syncthreads();

  unsigned prefix = 0, himask = 0;
  for (int pass = 0; pass < 4; ++pass) {
    const int shift = 24 - 8 * pass;
    hist[tid] = 0;                      // 256 threads, 256 bins
    __syncthreads();
    for (int i = tid; i < D_FF; i += 256) {
      unsigned u = keys[i];
      if ((u & himask) == prefix)
        atomicAdd(&hist[(u >> shift) & 255], 1);
    }
    __syncthreads();
    if (tid == 0) {
      int k = s_k[0], cum = 0, d = 255;
      for (; d >= 0; --d) {
        int c = hist[d];
        if (cum + c >= k) { s_k[0] = k - cum; break; }
        cum += c;
      }
      s_digit[0] = (d < 0) ? 0 : d;
    }
    __syncthreads();
    prefix |= ((unsigned)s_digit[0]) << (unsigned)shift;
    himask |= (0xFFu << (unsigned)shift);
  }

  // prefix == key of the k-th largest element; keep key >= prefix.
  __bf16* hrow = hb + (size_t)row * D_FF;
  const __bf16 zero = f32_to_bf16(0.0f);
  for (int i = tid; i < D_FF; i += 256) {
    if (keys[i] < prefix) hrow[i] = zero;
  }
}

// ---------------------------------------------------------------------------
// Kernel 3: down-projection GEMM  out[t][d] = sum_f h[t][f] * Wo[d][f].
// 32x32 tile per wave: 4 WMMA per K-step from 4 fragment loads, K = 4096.
// ---------------------------------------------------------------------------
__global__ __launch_bounds__(256)
void gemm_out(const __bf16* __restrict__ hb,
              const __bf16* __restrict__ wob,
              float* __restrict__ out) {
  const int lane = threadIdx.x & 31;
  const int wave = threadIdx.x >> 5;
  const int m0 = blockIdx.x * 32;
  const int d0 = (blockIdx.y * 8 + wave) * 32;
  const int r  = lane & 15;
  const int kh = lane >> 4;

  const __bf16* a0 = hb  + (size_t)(m0 + r) * D_FF + kh * 8;
  const __bf16* a1 = a0  + (size_t)16 * D_FF;
  const __bf16* b0 = wob + (size_t)(d0 + r) * D_FF + kh * 8;
  const __bf16* b1 = b0  + (size_t)16 * D_FF;

  v8f c[2][2] = {};
  for (int kk = 0; kk < D_FF; kk += 32) {
    __builtin_prefetch(a0 + kk + 128, 0, 1);
    __builtin_prefetch(b0 + kk + 128, 0, 1);
    v16bf A0 = load_frag(a0 + kk);
    v16bf A1 = load_frag(a1 + kk);
    v16bf B0 = load_frag(b0 + kk);
    v16bf B1 = load_frag(b1 + kk);
    c[0][0] = wmma_bf16(A0, B0, c[0][0]);
    c[0][1] = wmma_bf16(A0, B1, c[0][1]);
    c[1][0] = wmma_bf16(A1, B0, c[1][0]);
    c[1][1] = wmma_bf16(A1, B1, c[1][1]);
  }
  #pragma unroll
  for (int mi = 0; mi < 2; ++mi) {
    #pragma unroll
    for (int di = 0; di < 2; ++di) {
      #pragma unroll
      for (int i = 0; i < 8; ++i) {
        int m = m0 + mi * 16 + i + 8 * kh;
        int d = d0 + di * 16 + r;
        out[(size_t)m * D_MODEL + (size_t)d] = c[mi][di][i];
      }
    }
  }
}

// ---------------------------------------------------------------------------
// Host-side launcher (graph-capturable: kernel launches on `stream` only)
// ---------------------------------------------------------------------------
extern "C" void kernel_launch(void* const* d_in, const int* in_sizes, int n_in,
                              void* d_out, int out_size, void* d_ws, size_t ws_size,
                              hipStream_t stream) {
  (void)in_sizes; (void)n_in; (void)out_size; (void)ws_size;
  const float* x  = (const float*)d_in[0];   // [8192,1024]
  const float* W1 = (const float*)d_in[1];   // [4096,1024]
  const float* W2 = (const float*)d_in[2];   // [4096,1024]
  const float* Wo = (const float*)d_in[3];   // [1024,4096]
  // d_in[4] = k (scalar); fixed at compile time (TOPK)

  char* ws = (char*)d_ws;
  size_t o = 0;
  __bf16* xb  = (__bf16*)(ws + o); o += (size_t)NROWS * D_MODEL * 2;  // 16 MB
  __bf16* w1b = (__bf16*)(ws + o); o += (size_t)D_FF  * D_MODEL * 2;  //  8 MB
  __bf16* w2b = (__bf16*)(ws + o); o += (size_t)D_FF  * D_MODEL * 2;  //  8 MB
  __bf16* wob = (__bf16*)(ws + o); o += (size_t)D_MODEL * D_FF  * 2;  //  8 MB
  float*  act = (float*) (ws + o); o += (size_t)NROWS * D_FF * 4;     // 128 MB
  __bf16* hb  = (__bf16*)(ws + o);                                    //  64 MB

  const int nx = NROWS * D_MODEL, nw = D_FF * D_MODEL;
  cvt_f32_bf16<<<(nx + 255) / 256, 256, 0, stream>>>(x,  xb,  nx);
  cvt_f32_bf16<<<(nw + 255) / 256, 256, 0, stream>>>(W1, w1b, nw);
  cvt_f32_bf16<<<(nw + 255) / 256, 256, 0, stream>>>(W2, w2b, nw);
  cvt_f32_bf16<<<(nw + 255) / 256, 256, 0, stream>>>(Wo, wob, nw);

  dim3 g1(NROWS / 32, D_FF / 256);      // 256 x 16 blocks, 8 waves each
  gemm_gate_act<<<g1, 256, 0, stream>>>(xb, w1b, w2b, act, hb);

  topk_mask<<<NROWS, 256, TOPK_SMEM_BYTES, stream>>>(act, hb);

  dim3 g3(NROWS / 32, D_MODEL / 256);   // 256 x 4 blocks
  gemm_out<<<g3, 256, 0, stream>>>(hb, wob, (float*)d_out);
}
